// QModel_65481071410714
// MI455X (gfx1250) — compile-verified, matrix operand order checked by reading, not run on hardware
//
#include <hip/hip_runtime.h>
#include <hip/hip_bf16.h>

#define NQ 14
#define NSTATE 16384      // 2^14
#define NLAYERS 4
#define BATCH 128
#define K1_THREADS 256

typedef __attribute__((ext_vector_type(16))) __bf16 v16bf;
typedef __attribute__((ext_vector_type(8)))  float  v8f;

// Composed permutation of the CNOT ring CNOT(0,1)..CNOT(12,13),CNOT(13,0).
// Wire i lives at bit (13-i). Sequential CNOTs give new_wire_i = XOR(wires 0..i)
// for i>=1 and new_wire_0 = XOR(wires 1..13). In bit space that is a suffix-XOR:
// y bit j = XOR of k bits j..13.
__device__ __forceinline__ int ring_perm(int k) {
    int y = k;
    y ^= y >> 1; y ^= y >> 2; y ^= y >> 4; y ^= y >> 8;
    // wires 1..13 (bits 12..0) come from y; wire 0 (bit 13) = (y&1) ^ bit13(k)
    return (y & 0x1FFF) | (((y ^ (k >> 13)) & 1) << 13);
}

// ---------------------------------------------------------------------------
// Kernel 1: one workgroup == one batch sample. Double-buffered 2^14 complex
// state in LDS (2 x 128 KB of the WGP's 320 KB). Per layer: 13 in-place fused
// RX*RZ passes + 1 pass fusing the wire-13 gate with the full CNOT-ring
// permutation scatter into the other buffer. float2 layout -> b64 DS ops.
// ---------------------------------------------------------------------------
__global__ void __launch_bounds__(K1_THREADS)
qsim_state_kernel(const float* __restrict__ states,
                  const float* __restrict__ weights,
                  unsigned short* __restrict__ pbuf) {
    extern __shared__ float2 dynlds[];
    float2* cur = dynlds;           // 16384 float2 (128 KB)
    float2* nxt = dynlds + NSTATE;  // 16384 float2 (128 KB)

    __shared__ float encC[NQ], encS[NQ];
    __shared__ float wCos[NLAYERS * NQ * 2];
    __shared__ float wSin[NLAYERS * NQ * 2];

    const int tid = threadIdx.x;
    const int b   = blockIdx.x;

    if (tid < NQ) {
        float x = fabsf(states[(size_t)b * NSTATE + tid]) * 0.5f;
        encC[tid] = cosf(x);
        encS[tid] = sinf(x);
    }
    if (tid < NLAYERS * NQ * 2) {
        float w = weights[tid] * 0.5f;
        wCos[tid] = cosf(w);
        wSin[tid] = sinf(w);
    }
    __syncthreads();

    // Product-state init: RX-encoded |0..0> is separable.
    // amp(k) = (-i)^popcount(k) * prod_i (bit_i ? sin_i : cos_i)
    for (int k = tid; k < NSTATE; k += K1_THREADS) {
        float mag = 1.0f;
        #pragma unroll
        for (int i = 0; i < NQ; ++i)
            mag *= ((k >> (NQ - 1 - i)) & 1) ? encS[i] : encC[i];
        int ph = __popc(k) & 3;
        float re = 0.0f, im = 0.0f;
        if      (ph == 0) re =  mag;
        else if (ph == 1) im = -mag;
        else if (ph == 2) re = -mag;
        else              im =  mag;
        cur[k] = make_float2(re, im);
    }
    __syncthreads();

    for (int l = 0; l < NLAYERS; ++l) {
        // Wires 0..12: fused RX*RZ, in place.
        for (int i = 0; i < NQ - 1; ++i) {
            const int base = (l * NQ + i) * 2;
            const float cx = wCos[base + 0], sx = wSin[base + 0];
            const float cz = wCos[base + 1], sz = wSin[base + 1];
            const int p = NQ - 1 - i;
            const int mask = 1 << p;
            for (int t = tid; t < (NSTATE >> 1); t += K1_THREADS) {
                int k0 = ((t & ~(mask - 1)) << 1) | (t & (mask - 1));
                int k1 = k0 | mask;
                float2 a0 = cur[k0], a1 = cur[k1];
                // RX
                float nr0 = cx * a0.x + sx * a1.y;
                float ni0 = cx * a0.y - sx * a1.x;
                float nr1 = cx * a1.x + sx * a0.y;
                float ni1 = cx * a1.y - sx * a0.x;
                // RZ
                cur[k0] = make_float2(cz * nr0 + sz * ni0, cz * ni0 - sz * nr0);
                cur[k1] = make_float2(cz * nr1 - sz * ni1, cz * ni1 + sz * nr1);
            }
            __syncthreads();
        }
        // Wire 13: fused RX*RZ + full CNOT-ring permutation scatter into nxt.
        {
            const int base = (l * NQ + (NQ - 1)) * 2;
            const float cx = wCos[base + 0], sx = wSin[base + 0];
            const float cz = wCos[base + 1], sz = wSin[base + 1];
            for (int t = tid; t < (NSTATE >> 1); t += K1_THREADS) {
                int k0 = 2 * t, k1 = 2 * t + 1;   // mask == 1
                float2 a0 = cur[k0], a1 = cur[k1];
                float nr0 = cx * a0.x + sx * a1.y;
                float ni0 = cx * a0.y - sx * a1.x;
                float nr1 = cx * a1.x + sx * a0.y;
                float ni1 = cx * a1.y - sx * a0.x;
                float2 o0 = make_float2(cz * nr0 + sz * ni0, cz * ni0 - sz * nr0);
                float2 o1 = make_float2(cz * nr1 - sz * ni1, cz * ni1 + sz * nr1);
                nxt[ring_perm(k0)] = o0;
                nxt[ring_perm(k1)] = o1;
            }
            __syncthreads();
            float2* tmp = cur; cur = nxt; nxt = tmp;
        }
    }

    // |psi|^2 -> bf16 (RNE) to workspace for the WMMA reduction.
    unsigned short* prow = pbuf + (size_t)b * NSTATE;
    for (int k = tid; k < NSTATE; k += K1_THREADS) {
        float2 a = cur[k];
        float pv = a.x * a.x + a.y * a.y;
        unsigned bits = __float_as_uint(pv);
        bits += 0x7FFFu + ((bits >> 16) & 1u);
        prow[k] = (unsigned short)(bits >> 16);
    }
}

// ---------------------------------------------------------------------------
// Kernel 2: Z(128x14) = P(128x16384) x S(16384x14) as bf16 WMMA GEMM.
// S[k,n] = (-1)^{bit_{13-n}(k)}; per-lane B column built branch-free:
// loop-invariant packed constant XOR a per-tile sign-bit splat.
// ---------------------------------------------------------------------------
__global__ void __launch_bounds__(32)
wmma_zhead_kernel(const unsigned short* __restrict__ pbuf,
                  const float* __restrict__ head_w,
                  const float* __restrict__ head_b,
                  float* __restrict__ out) {
    __shared__ float ztile[16][16];

    const int lane = threadIdx.x;
    const int n    = lane & 15;   // A row within tile == B column
    const int h    = lane >> 4;

    // Build loop-invariant packed-bf16 B constants and per-tile flip mask.
    unsigned short elem[16];
    #pragma unroll
    for (int j = 0; j < 16; ++j) {
        int koff = (j < 8) ? (8 * h + j) : (16 + 8 * h + (j - 8));
        unsigned short v;
        if (n >= NQ)      v = 0x0000u;                                    // pad col
        else if (n >= 9)  v = ((koff >> (NQ - 1 - n)) & 1) ? 0xBF80u : 0x3F80u;
        else              v = 0x3F80u;                                    // +1, sign from tile
        elem[j] = v;
    }
    unsigned cb[8];
    #pragma unroll
    for (int j = 0; j < 8; ++j) cb[j] = (unsigned)elem[2 * j] | ((unsigned)elem[2 * j + 1] << 16);
    const unsigned tmask = (n <= 8) ? (1u << (8 - n)) : 0u;   // bit of kt giving sign

    const int srow = blockIdx.x * 16 + n;
    const unsigned short* prow = pbuf + (size_t)srow * NSTATE;

    v8f acc = {};
    for (int kt = 0; kt < NSTATE / 32; ++kt) {
        const int kbase = kt * 32;
        const unsigned flip = ((unsigned)kt & tmask) ? 0x80008000u : 0u;
        union { unsigned u[8]; v16bf v; } B;
        #pragma unroll
        for (int j = 0; j < 8; ++j) B.u[j] = cb[j] ^ flip;

        union { uint4 q; __bf16 e[8]; } c0, c1;
        c0.q = *(const uint4*)(prow + kbase + 8 * h);
        c1.q = *(const uint4*)(prow + kbase + 16 + 8 * h);
        v16bf avec;
        #pragma unroll
        for (int j = 0; j < 8; ++j) { avec[j] = c0.e[j]; avec[8 + j] = c1.e[j]; }

        acc = __builtin_amdgcn_wmma_f32_16x16x32_bf16(
            false, avec, false, B.v, (short)0, acc, false, false);
    }

    #pragma unroll
    for (int j = 0; j < 8; ++j) ztile[j + 8 * h][n] = acc[j];
    __syncthreads();

    if (lane < 16) {
        const int bi = blockIdx.x * 16 + lane;
        float o = head_b[0];
        #pragma unroll
        for (int w = 0; w < NQ; ++w) o += ztile[lane][w] * head_w[w];
        out[bi] = o;
    }
}

extern "C" void kernel_launch(void* const* d_in, const int* in_sizes, int n_in,
                              void* d_out, int out_size, void* d_ws, size_t ws_size,
                              hipStream_t stream) {
    const float* states  = (const float*)d_in[0];   // (128, 16384) f32
    const float* weights = (const float*)d_in[1];   // (4, 14, 2)   f32
    const float* head_w  = (const float*)d_in[2];   // (1, 14)      f32
    const float* head_b  = (const float*)d_in[3];   // (1,)         f32
    float* out = (float*)d_out;                     // (128,)       f32
    unsigned short* pbuf = (unsigned short*)d_ws;   // 128*16384 bf16 = 4 MB

    // Kernel 1: 128 WGs (one per sample), 8 wave32s each, 256 KB dynamic LDS
    // (double-buffered state; CDNA5 WGP has 320 KB).
    qsim_state_kernel<<<BATCH, K1_THREADS, 2 * NSTATE * sizeof(float2), stream>>>(
        states, weights, pbuf);

    // Kernel 2: 8 M-tiles of 16 samples, one wave each.
    wmma_zhead_kernel<<<BATCH / 16, 32, 0, stream>>>(pbuf, head_w, head_b, out);
}
